// GraphInteractionLayer_52304111730731
// MI455X (gfx1250) — matrix operand order repaired and checked by reading, not compiled, not run on hardware
//
#include <hip/hip_runtime.h>
#include <hip/hip_bf16.h>

// ---------------------------------------------------------------------------
// Types for CDNA5 WMMA intrinsics
// ---------------------------------------------------------------------------
typedef __bf16    bf16_t;
typedef __bf16    v8bf  __attribute__((ext_vector_type(8)));
typedef __bf16    v16bf __attribute__((ext_vector_type(16)));
typedef _Float16  v8h   __attribute__((ext_vector_type(8)));
typedef _Float16  v16h  __attribute__((ext_vector_type(16)));
typedef float     v8f   __attribute__((ext_vector_type(8)));

__device__ __forceinline__ bf16_t f2bf(float f) {
  unsigned u = __builtin_bit_cast(unsigned, f);
  unsigned r = (u + 0x7fffu + ((u >> 16) & 1u)) >> 16;
  unsigned short s = (unsigned short)r;
  return __builtin_bit_cast(bf16_t, s);
}

__device__ __forceinline__ v16bf cat16bf(v8bf lo, v8bf hi) {
  return __builtin_shufflevector(lo, hi, 0,1,2,3,4,5,6,7,8,9,10,11,12,13,14,15);
}
__device__ __forceinline__ v16h cat16h(v8h lo, v8h hi) {
  return __builtin_shufflevector(lo, hi, 0,1,2,3,4,5,6,7,8,9,10,11,12,13,14,15);
}

// ---------------------------------------------------------------------------
// Elementwise f32 -> bf16 cast
// ---------------------------------------------------------------------------
__global__ __launch_bounds__(256)
void cast_bf16_kernel(const float* __restrict__ in, bf16_t* __restrict__ out, int n) {
  int i = blockIdx.x * 256 + threadIdx.x;
  if (i < n) out[i] = f2bf(in[i]);
}

// ---------------------------------------------------------------------------
// Epilogue: store one 16x16 f32 accumulator tile (C/D layout per ISA 7.12.2:
// element r -> M = r + 8*(lane>=16), N = lane%16).
// ---------------------------------------------------------------------------
__device__ __forceinline__ void store_tile(const v8f& acc, void* __restrict__ out,
                                           const float* __restrict__ bias,
                                           int mbase, int nbase, int ldc, size_t zoff,
                                           float scale, int relu, int mode, int lane) {
  const int mr0 = (lane & 16) >> 1;
  const int nc  = lane & 15;
  const float bval = bias ? bias[nbase + nc] : 0.f;
#pragma unroll
  for (int r = 0; r < 8; ++r) {
    float val = acc[r] * scale + bval;
    if (relu) val = fmaxf(val, 0.f);
    size_t idx = zoff + (size_t)(mbase + r + mr0) * ldc + (nbase + nc);
    if (mode == 0)      ((float*)out)[idx]    = val;
    else if (mode == 1) ((_Float16*)out)[idx] = (_Float16)val;
    else                ((bf16_t*)out)[idx]   = f2bf(val);
  }
}

// ---------------------------------------------------------------------------
// bf16 WMMA GEMM:  C[z, m, n] = scale * sum_k A[z,m,k] * Bt[z,n,k]
//                  (+ bias[n]) (+ relu), out mode: 0=f32 1=f16 2=bf16
// Block = 128 threads = 4 waves arranged 2x2; each wave computes a 32x32
// output patch with 2x2 register blocking: per k-step, 2 A-frags + 2 B-frags
// (8 global_load_b128) feed 4 independent v_wmma_f32_16x16x32_bf16 — 2x the
// arithmetic intensity of one-tile-per-wave, and the 4 accumulators hide the
// WMMA->WMMA result latency. Fragment layouts per ISA 7.12.2.
// ---------------------------------------------------------------------------
__global__ __launch_bounds__(128)
void gemm_bf16_kernel(const bf16_t* __restrict__ A, const bf16_t* __restrict__ Bt,
                      const float* __restrict__ bias, void* __restrict__ out,
                      int M, int N, int K, int lda, int ldb, int ldc,
                      long long sA, long long sB, long long sC,
                      float scale, int relu, int mode) {
  (void)M; (void)N;
  const int lane = threadIdx.x & 31;
  const int wave = threadIdx.x >> 5;
  const int m0 = blockIdx.y * 64 + (wave >> 1) * 32;
  const int n0 = blockIdx.x * 64 + (wave & 1) * 32;
  const int z  = blockIdx.z;

  const bf16_t* Ab = A  + (size_t)z * sA;
  const bf16_t* Bb = Bt + (size_t)z * sB;

  const int mn  = lane & 15;          // row of A frag / row (=n) of Bt frag
  const int kbA = (lane & 16) >> 1;   // 0 or 8  (A: upper half-wave holds K+8)
  const int kbB = (lane & 16) >> 1;   // 0 or 8  (B: upper half-wave holds K+8)
  const bf16_t* pa0 = Ab + (size_t)(m0 + mn) * lda + kbA;
  const bf16_t* pa1 = pa0 + (size_t)16 * lda;
  const bf16_t* pb0 = Bb + (size_t)(n0 + mn) * ldb + kbB;
  const bf16_t* pb1 = pb0 + (size_t)16 * ldb;

  v8f acc00 = {0.f,0.f,0.f,0.f,0.f,0.f,0.f,0.f};
  v8f acc01 = acc00, acc10 = acc00, acc11 = acc00;

  for (int k0 = 0; k0 < K; k0 += 32) {
    __builtin_prefetch(pa0 + k0 + 128, 0, 1);
    __builtin_prefetch(pb0 + k0 + 128, 0, 1);
    v16bf a0 = cat16bf(*(const v8bf*)(pa0 + k0), *(const v8bf*)(pa0 + k0 + 16));
    v16bf a1 = cat16bf(*(const v8bf*)(pa1 + k0), *(const v8bf*)(pa1 + k0 + 16));
    v16bf b0 = cat16bf(*(const v8bf*)(pb0 + k0), *(const v8bf*)(pb0 + k0 + 16));
    v16bf b1 = cat16bf(*(const v8bf*)(pb1 + k0), *(const v8bf*)(pb1 + k0 + 16));
    acc00 = __builtin_amdgcn_wmma_f32_16x16x32_bf16(false, a0, false, b0,
                                                    (short)0, acc00, false, false);
    acc01 = __builtin_amdgcn_wmma_f32_16x16x32_bf16(false, a0, false, b1,
                                                    (short)0, acc01, false, false);
    acc10 = __builtin_amdgcn_wmma_f32_16x16x32_bf16(false, a1, false, b0,
                                                    (short)0, acc10, false, false);
    acc11 = __builtin_amdgcn_wmma_f32_16x16x32_bf16(false, a1, false, b1,
                                                    (short)0, acc11, false, false);
  }

  const size_t zoff = (size_t)z * sC;
  store_tile(acc00, out, bias, m0 +  0, n0 +  0, ldc, zoff, scale, relu, mode, lane);
  store_tile(acc01, out, bias, m0 +  0, n0 + 16, ldc, zoff, scale, relu, mode, lane);
  store_tile(acc10, out, bias, m0 + 16, n0 +  0, ldc, zoff, scale, relu, mode, lane);
  store_tile(acc11, out, bias, m0 + 16, n0 + 16, ldc, zoff, scale, relu, mode, lane);
}

// ---------------------------------------------------------------------------
// Pairwise-position bias MLP + diag mask + row softmax -> attn (f16)
// One block (256 threads) per attention row (b, n); MLP weights in LDS.
// ---------------------------------------------------------------------------
__global__ __launch_bounds__(256)
void bias_softmax_kernel(const float* __restrict__ logits, const float* __restrict__ centers,
                         const float* __restrict__ Wr1, const float* __restrict__ br1,
                         const float* __restrict__ Wr2, const float* __restrict__ br2,
                         _Float16* __restrict__ attn, int Nn) {
  __shared__ float w0[512], w1[512], cb[512], w2[512];
  __shared__ float red[256];
  const int m = threadIdx.x;
  const int n = blockIdx.x;
  const int b = blockIdx.y;

  for (int i = m; i < 512; i += 256) {
    w0[i] = Wr1[2 * i];
    w1[i] = Wr1[2 * i + 1];
    cb[i] = br1[i];
    w2[i] = Wr2[i];
  }
  __syncthreads();

  const float dx = centers[((size_t)b * Nn + n) * 2 + 0] - centers[((size_t)b * Nn + m) * 2 + 0];
  const float dy = centers[((size_t)b * Nn + n) * 2 + 1] - centers[((size_t)b * Nn + m) * 2 + 1];
  float s = 0.f;
#pragma unroll 8
  for (int h = 0; h < 512; ++h) {
    float t = fmaf(dx, w0[h], fmaf(dy, w1[h], cb[h]));
    t = fmaxf(t, 0.f);
    s = fmaf(t, w2[h], s);
  }

  float logit = logits[((size_t)b * Nn + n) * Nn + m] + s + br2[0];
  if (m == n) logit = -__builtin_inff();

  red[m] = logit; __syncthreads();
#pragma unroll
  for (int st = 128; st > 0; st >>= 1) {
    if (m < st) red[m] = fmaxf(red[m], red[m + st]);
    __syncthreads();
  }
  const float mx = red[0]; __syncthreads();
  const float e = __expf(logit - mx);
  red[m] = e; __syncthreads();
#pragma unroll
  for (int st = 128; st > 0; st >>= 1) {
    if (m < st) red[m] += red[m + st];
    __syncthreads();
  }
  attn[((size_t)b * Nn + n) * Nn + m] = (_Float16)(e / red[0]);
}

// ---------------------------------------------------------------------------
// agg[b,n,d] = sum_m attn[b,n,m] * v[b,m,d]   via f16 WMMA.
// V tile is transposed through LDS (Bt[d][m] layout, padded pitch 40 halves
// so fragment reads stay 16B aligned -> ds_load_b128).
// ---------------------------------------------------------------------------
__global__ __launch_bounds__(128)
void attn_v_kernel(const _Float16* __restrict__ attn, const _Float16* __restrict__ v,
                   bf16_t* __restrict__ agg, int Nn, int Dd) {
  __shared__ _Float16 vt[64][40];
  const int lane = threadIdx.x & 31;
  const int wave = threadIdx.x >> 5;
  const int b  = blockIdx.z;
  const int m0 = blockIdx.y * 16;   // output rows (query index n)
  const int d0 = blockIdx.x * 64;   // output cols (feature d), 16 per wave

  const _Float16* Ab = attn + (size_t)b * Nn * Nn;
  const _Float16* Vb = v    + (size_t)b * Nn * Dd;

  const int mn  = lane & 15;
  const int kbA = (lane & 16) >> 1;
  const int kbB = (lane & 16) >> 1;
  const _Float16* pa = Ab + (size_t)(m0 + mn) * Nn + kbA;
  const _Float16* pb = &vt[wave * 16 + mn][kbB];

  v8f acc = {0.f,0.f,0.f,0.f,0.f,0.f,0.f,0.f};
  for (int k0 = 0; k0 < Nn; k0 += 32) {
    for (int e = threadIdx.x; e < 64 * 32; e += 128) {
      int mm = e >> 6, dd = e & 63;
      vt[dd][mm] = Vb[(size_t)(k0 + mm) * Dd + d0 + dd];
    }
    __syncthreads();
    v16h af = cat16h(*(const v8h*)(pa + k0), *(const v8h*)(pa + k0 + 16));
    v16h bf = cat16h(*(const v8h*)pb, *(const v8h*)(pb + 16));
    acc = __builtin_amdgcn_wmma_f32_16x16x32_f16(false, af, false, bf,
                                                 (short)0, acc, false, false);
    __syncthreads();
  }

  const int mr0 = (lane & 16) >> 1;
  const int nc  = lane & 15;
#pragma unroll
  for (int r = 0; r < 8; ++r) {
    agg[((size_t)b * Nn + m0 + r + mr0) * Dd + d0 + wave * 16 + nc] = f2bf(acc[r]);
  }
}

// ---------------------------------------------------------------------------
// y = LayerNorm(res + delta) * g + b ; writes f32 and optional bf16 copy.
// One block (128 threads) per token (D = 512 -> 4 elems/thread).
// ---------------------------------------------------------------------------
__global__ __launch_bounds__(128)
void add_ln_kernel(const float* __restrict__ res, const float* __restrict__ dlt,
                   const float* __restrict__ g, const float* __restrict__ bb,
                   float* __restrict__ out, bf16_t* __restrict__ out_bf) {
  __shared__ float s1[128], s2[128];
  const int t = blockIdx.x, tid = threadIdx.x;
  const float* rp = res + (size_t)t * 512;
  const float* dp = dlt + (size_t)t * 512;
  float v[4], sum = 0.f, sq = 0.f;
#pragma unroll
  for (int j = 0; j < 4; ++j) {
    int i = tid + j * 128;
    float x = rp[i] + dp[i];
    v[j] = x; sum += x; sq += x * x;
  }
  s1[tid] = sum; s2[tid] = sq; __syncthreads();
#pragma unroll
  for (int st = 64; st > 0; st >>= 1) {
    if (tid < st) { s1[tid] += s1[tid + st]; s2[tid] += s2[tid + st]; }
    __syncthreads();
  }
  const float mu  = s1[0] * (1.f / 512.f);
  const float var = s2[0] * (1.f / 512.f) - mu * mu;
  const float inv = rsqrtf(var + 1e-5f);
#pragma unroll
  for (int j = 0; j < 4; ++j) {
    int i = tid + j * 128;
    float y = (v[j] - mu) * inv * g[i] + bb[i];
    out[(size_t)t * 512 + i] = y;
    if (out_bf) out_bf[(size_t)t * 512 + i] = f2bf(y);
  }
}

// ---------------------------------------------------------------------------
// Host launcher
// ---------------------------------------------------------------------------
extern "C" void kernel_launch(void* const* d_in, const int* in_sizes, int n_in,
                              void* d_out, int out_size, void* d_ws, size_t ws_size,
                              hipStream_t stream) {
  (void)in_sizes; (void)n_in; (void)out_size; (void)ws_size;
  const float* x    = (const float*)d_in[0];
  const float* ctr  = (const float*)d_in[1];
  const float* Wq   = (const float*)d_in[2];
  const float* bq   = (const float*)d_in[3];
  const float* Wk   = (const float*)d_in[4];
  const float* bk   = (const float*)d_in[5];
  const float* Wv   = (const float*)d_in[6];
  const float* bv   = (const float*)d_in[7];
  const float* Wr1  = (const float*)d_in[8];
  const float* br1  = (const float*)d_in[9];
  const float* Wr2  = (const float*)d_in[10];
  const float* br2  = (const float*)d_in[11];
  const float* Wo   = (const float*)d_in[12];
  const float* bo   = (const float*)d_in[13];
  const float* g1   = (const float*)d_in[14];
  const float* b1   = (const float*)d_in[15];
  const float* Wf1  = (const float*)d_in[16];
  const float* bf1  = (const float*)d_in[17];
  const float* Wf2  = (const float*)d_in[18];
  const float* bf2  = (const float*)d_in[19];
  const float* g2   = (const float*)d_in[20];
  const float* b2   = (const float*)d_in[21];

  const int B = 8, N = 256, D = 512, T = B * N;  // 2048 tokens
  char* base = (char*)d_ws;
  size_t off = 0;
  auto alloc = [&](size_t bytes) -> void* {
    off = (off + 255) & ~(size_t)255;
    void* p = base + off;
    off += bytes;
    return p;
  };

  bf16_t*   xb      = (bf16_t*)  alloc((size_t)T * D * 2);
  bf16_t*   Wqb     = (bf16_t*)  alloc((size_t)D * D * 2);
  bf16_t*   Wkb     = (bf16_t*)  alloc((size_t)D * D * 2);
  bf16_t*   Wvb     = (bf16_t*)  alloc((size_t)D * D * 2);
  bf16_t*   Wob     = (bf16_t*)  alloc((size_t)D * D * 2);
  bf16_t*   Wf1b    = (bf16_t*)  alloc((size_t)2 * D * D * 2);
  bf16_t*   Wf2b    = (bf16_t*)  alloc((size_t)2 * D * D * 2);
  bf16_t*   qb      = (bf16_t*)  alloc((size_t)T * D * 2);
  bf16_t*   kb      = (bf16_t*)  alloc((size_t)T * D * 2);
  _Float16* vh      = (_Float16*)alloc((size_t)T * D * 2);
  float*    logits  = (float*)   alloc((size_t)B * N * N * 4);
  _Float16* attnh   = (_Float16*)alloc((size_t)B * N * N * 2);
  bf16_t*   aggb    = (bf16_t*)  alloc((size_t)T * D * 2);
  float*    attnout = (float*)   alloc((size_t)T * D * 4);
  float*    x1      = (float*)   alloc((size_t)T * D * 4);
  bf16_t*   x1b     = (bf16_t*)  alloc((size_t)T * D * 2);
  bf16_t*   fhb     = (bf16_t*)  alloc((size_t)T * 2 * D * 2);
  float*    f2      = (float*)   alloc((size_t)T * D * 4);

  // -- cast inputs/weights to bf16 --------------------------------------------
  cast_bf16_kernel<<<(T * D + 255) / 256, 256, 0, stream>>>(x,   xb,   T * D);
  cast_bf16_kernel<<<(D * D + 255) / 256, 256, 0, stream>>>(Wq,  Wqb,  D * D);
  cast_bf16_kernel<<<(D * D + 255) / 256, 256, 0, stream>>>(Wk,  Wkb,  D * D);
  cast_bf16_kernel<<<(D * D + 255) / 256, 256, 0, stream>>>(Wv,  Wvb,  D * D);
  cast_bf16_kernel<<<(D * D + 255) / 256, 256, 0, stream>>>(Wo,  Wob,  D * D);
  cast_bf16_kernel<<<(2 * D * D + 255) / 256, 256, 0, stream>>>(Wf1, Wf1b, 2 * D * D);
  cast_bf16_kernel<<<(2 * D * D + 255) / 256, 256, 0, stream>>>(Wf2, Wf2b, 2 * D * D);

  // -- Q, K (bf16), V (f16) projections ---------------------------------------
  dim3 gQKV(D / 64, T / 64, 1);
  gemm_bf16_kernel<<<gQKV, 128, 0, stream>>>(xb, Wqb, bq, qb, T, D, D, D, D, D,
                                             0, 0, 0, 1.f, 0, 2);
  gemm_bf16_kernel<<<gQKV, 128, 0, stream>>>(xb, Wkb, bk, kb, T, D, D, D, D, D,
                                             0, 0, 0, 1.f, 0, 2);
  gemm_bf16_kernel<<<gQKV, 128, 0, stream>>>(xb, Wvb, bv, vh, T, D, D, D, D, D,
                                             0, 0, 0, 1.f, 0, 1);

  // -- logits = q @ k^T / sqrt(D), batched over B -----------------------------
  dim3 gLog(N / 64, N / 64, B);
  gemm_bf16_kernel<<<gLog, 128, 0, stream>>>(qb, kb, nullptr, logits,
                                             N, N, D, D, D, N,
                                             (long long)N * D, (long long)N * D,
                                             (long long)N * N,
                                             0.044194173824159216f /*1/sqrt(512)*/, 0, 0);

  // -- relative-position bias MLP + mask + softmax -> attn (f16) --------------
  bias_softmax_kernel<<<dim3(N, B), 256, 0, stream>>>(logits, ctr, Wr1, br1, Wr2,
                                                      br2, attnh, N);

  // -- agg = attn @ v (f16 WMMA, LDS-transposed V) ----------------------------
  attn_v_kernel<<<dim3(D / 64, N / 16, B), 128, 0, stream>>>(attnh, vh, aggb, N, D);

  // -- attnout = agg @ Wo^T + bo ----------------------------------------------
  gemm_bf16_kernel<<<gQKV, 128, 0, stream>>>(aggb, Wob, bo, attnout, T, D, D,
                                             D, D, D, 0, 0, 0, 1.f, 0, 0);

  // -- x1 = LN(x + attnout) ---------------------------------------------------
  add_ln_kernel<<<T, 128, 0, stream>>>(x, attnout, g1, b1, x1, x1b);

  // -- FFN: fh = relu(x1 @ Wf1^T + bf1); f2 = fh @ Wf2^T + bf2 ----------------
  gemm_bf16_kernel<<<dim3(2 * D / 64, T / 64, 1), 128, 0, stream>>>(
      x1b, Wf1b, bf1, fhb, T, 2 * D, D, D, D, 2 * D, 0, 0, 0, 1.f, 1, 2);
  gemm_bf16_kernel<<<dim3(D / 64, T / 64, 1), 128, 0, stream>>>(
      fhb, Wf2b, bf2, f2, T, D, 2 * D, 2 * D, 2 * D, D, 0, 0, 0, 1.f, 0, 0);

  // -- out = LN(x1 + f2) ------------------------------------------------------
  add_ln_kernel<<<T, 128, 0, stream>>>(x1, f2, g2, b2, (float*)d_out, nullptr);
}
